// StyleLoss_22239340659533
// MI455X (gfx1250) — matrix operand order repaired and checked by reading, not compiled, not run on hardware
//
#include <hip/hip_runtime.h>
#include <hip/hip_bf16.h>

typedef __bf16 bf16_t;
typedef __attribute__((ext_vector_type(16))) __bf16 v16bf;
typedef __attribute__((ext_vector_type(8)))  __bf16 v8bf;
typedef __attribute__((ext_vector_type(8)))  float  v8f;
typedef __attribute__((ext_vector_type(2)))  float  v2f;

#define D_DIM 512
#define N_DIM 4096

// ---------- helpers ----------
__device__ __forceinline__ unsigned f2ord(float f) {
  unsigned b = __float_as_uint(f);
  return (b & 0x80000000u) ? ~b : (b | 0x80000000u);
}
__device__ __forceinline__ float ord2f(unsigned k) {
  unsigned b = (k & 0x80000000u) ? (k ^ 0x80000000u) : ~k;
  return __uint_as_float(b);
}
__device__ __forceinline__ float wave_sum(float v) {
#pragma unroll
  for (int m = 16; m >= 1; m >>= 1) v += __shfl_xor(v, m, 32);
  return v;
}

// A fragment: 16x32 bf16, row-major source [row][stride], K contiguous.
// Lane L (hi=L>=16, m=L&15): elems 0..7 -> K = 8*hi + 0..7 ; elems 8..15 -> K = 8*hi + 16..23
__device__ __forceinline__ v16bf load_fragA(const bf16_t* __restrict__ base, int row, int stride,
                                            int k0, int lane) {
  int hi = lane >> 4, m = lane & 15;
  const bf16_t* p = base + (size_t)(row + m) * stride + k0 + 8 * hi;
  v8bf c0 = *(const v8bf*)p;
  v8bf c1 = *(const v8bf*)(p + 16);
  v16bf a;
#pragma unroll
  for (int i = 0; i < 8; ++i) { a[i] = c0[i]; a[8 + i] = c1[i]; }
  return a;
}
// B fragment: 32x16 bf16; B[k][j] sourced from rows of the transposed operand (col j, K contiguous).
// Lane L (hi, n=L&15): elems 0..7 -> K = 8*hi + 0..7 ; elems 8..15 -> K = 8*hi + 16..23
__device__ __forceinline__ v16bf load_fragB(const bf16_t* __restrict__ base, int col, int stride,
                                            int k0, int lane) {
  int hi = lane >> 4, n = lane & 15;
  const bf16_t* p = base + (size_t)(col + n) * stride + k0 + 8 * hi;
  v8bf c0 = *(const v8bf*)p;
  v8bf c1 = *(const v8bf*)(p + 16);
  v16bf b;
#pragma unroll
  for (int i = 0; i < 8; ++i) { b[i] = c0[i]; b[8 + i] = c1[i]; }
  return b;
}
__device__ __forceinline__ v8f wmma_bf16(v16bf a, v16bf b, v8f c) {
  return __builtin_amdgcn_wmma_f32_16x16x32_bf16(false, a, false, b, (short)0, c, false, false);
}

// ---------- prep kernels ----------
__global__ void k_init(unsigned* __restrict__ keys, float* __restrict__ scalars) {
  for (int i = threadIdx.x; i < 4 * N_DIM; i += blockDim.x) keys[i] = 0xFF800000u;  // ord(+inf)
  if (threadIdx.x == 0) { scalars[0] = 0.f; scalars[1] = 0.f; }
}

// fp32 [512][4096] -> bf16 transposed [4096][512] + per-column L2 norm
__global__ void k_convert(const float* __restrict__ src, bf16_t* __restrict__ T,
                          float* __restrict__ norms) {
  int n = blockIdx.x * blockDim.x + threadIdx.x;
  float ss = 0.f;
  for (int c = 0; c < D_DIM; ++c) {
    float v = src[(size_t)c * N_DIM + n];
    ss += v * v;
    T[(size_t)n * D_DIM + c] = (bf16_t)v;
  }
  norms[n] = sqrtf(ss);
}

__global__ void k_chan_mean(const float* __restrict__ src, float* __restrict__ mean) {
  __shared__ float red[256];
  int c = blockIdx.x;
  float s = 0.f;
  for (int n = threadIdx.x; n < N_DIM; n += 256) s += src[(size_t)c * N_DIM + n];
  red[threadIdx.x] = s;
  __syncthreads();
  for (int st = 128; st > 0; st >>= 1) {
    if (threadIdx.x < st) red[threadIdx.x] += red[threadIdx.x + st];
    __syncthreads();
  }
  if (threadIdx.x == 0) mean[c] = red[0] / (float)N_DIM;
}

__global__ void k_center(const float* __restrict__ src, const float* __restrict__ mean,
                         bf16_t* __restrict__ out) {
  size_t i = (size_t)blockIdx.x * blockDim.x + threadIdx.x;
  int c = (int)(i >> 12);  // / 4096
  out[i] = (bf16_t)(src[i] - mean[c]);
}

// RGB (first 3 channels) -> YUV, padded to 4 floats/row, plus 3-d norms
__global__ void k_yuv(const float* __restrict__ src, float* __restrict__ uvT,
                      float* __restrict__ n3) {
  int n = blockIdx.x * blockDim.x + threadIdx.x;
  float r = src[n], g = src[N_DIM + n], b = src[2 * N_DIM + n];
  float y0 =  0.577350f * r + 0.577350f * g + 0.577350f * b;
  float y1 = -0.577350f * r + 0.788675f * g - 0.211325f * b;
  float y2 = -0.577350f * r - 0.211325f * g + 0.788675f * b;
  uvT[(size_t)n * 4 + 0] = y0;
  uvT[(size_t)n * 4 + 1] = y1;
  uvT[(size_t)n * 4 + 2] = y2;
  uvT[(size_t)n * 4 + 3] = 0.f;
  n3[n] = sqrtf(y0 * y0 + y1 * y1 + y2 * y2);
}

// ---------- content loss: mean |cos(ef,ef) - cos(cf,cf)|, fused dual-Gram ----------
__global__ void __launch_bounds__(256) k_content(const bf16_t* __restrict__ Te,
                                                 const bf16_t* __restrict__ Tc,
                                                 const float* __restrict__ ne,
                                                 const float* __restrict__ nc,
                                                 float* __restrict__ out_sum) {
  int lane = threadIdx.x & 31, wave = threadIdx.x >> 5;
  int n0 = blockIdx.y * 64, m0 = blockIdx.x * 64;
  int rowA = n0 + 16 * (wave & 3);
  int colB0 = m0 + 16 * ((wave >> 2) * 2), colB1 = colB0 + 16;
  v8f ae0 = {}, ae1 = {}, ac0 = {}, ac1 = {};
  for (int k0 = 0; k0 < D_DIM; k0 += 32) {
    v16bf aE = load_fragA(Te, rowA, D_DIM, k0, lane);
    v16bf aC = load_fragA(Tc, rowA, D_DIM, k0, lane);
    v16bf bE0 = load_fragB(Te, colB0, D_DIM, k0, lane);
    v16bf bE1 = load_fragB(Te, colB1, D_DIM, k0, lane);
    v16bf bC0 = load_fragB(Tc, colB0, D_DIM, k0, lane);
    v16bf bC1 = load_fragB(Tc, colB1, D_DIM, k0, lane);
    ae0 = wmma_bf16(aE, bE0, ae0);
    ae1 = wmma_bf16(aE, bE1, ae1);
    ac0 = wmma_bf16(aC, bC0, ac0);
    ac1 = wmma_bf16(aC, bC1, ac1);
  }
  int hi = lane >> 4, jl = lane & 15;
  float local = 0.f;
#pragma unroll
  for (int t = 0; t < 2; ++t) {
    int jg = (t ? colB1 : colB0) + jl;
    float nej = ne[jg], ncj = nc[jg];
    const v8f& ge = t ? ae1 : ae0;
    const v8f& gc = t ? ac1 : ac0;
#pragma unroll
    for (int r = 0; r < 8; ++r) {
      int ig = rowA + r + 8 * hi;
      // |cx - cy| = |gc/(nc_i*nc_j) - ge/(ne_i*ne_j)|
      local += fabsf(gc[r] / (nc[ig] * ncj) - ge[r] / (ne[ig] * nej));
    }
  }
  local = wave_sum(local);
  if (lane == 0) atomicAdd(out_sum, local);
}

// ---------- REMD main: fused cosine-cost tiles + row/col min atomics ----------
__global__ void __launch_bounds__(256) k_cost_min(const bf16_t* __restrict__ Tx,
                                                  const bf16_t* __restrict__ Ty,
                                                  const float* __restrict__ nx,
                                                  const float* __restrict__ ny,
                                                  unsigned* __restrict__ rowkey,
                                                  unsigned* __restrict__ colkey) {
  int lane = threadIdx.x & 31, wave = threadIdx.x >> 5;
  int n0 = blockIdx.y * 64, m0 = blockIdx.x * 64;
  int rowA = n0 + 16 * (wave & 3);
  int colB0 = m0 + 16 * ((wave >> 2) * 2), colB1 = colB0 + 16;
  v8f a0 = {}, a1 = {};
  for (int k0 = 0; k0 < D_DIM; k0 += 32) {
    v16bf aX = load_fragA(Tx, rowA, D_DIM, k0, lane);
    v16bf b0 = load_fragB(Ty, colB0, D_DIM, k0, lane);
    v16bf b1 = load_fragB(Ty, colB1, D_DIM, k0, lane);
    a0 = wmma_bf16(aX, b0, a0);
    a1 = wmma_bf16(aX, b1, a1);
  }
  int hi = lane >> 4, jl = lane & 15;
#pragma unroll
  for (int t = 0; t < 2; ++t) {
    int jg = (t ? colB1 : colB0) + jl;
    const v8f& g = t ? a1 : a0;
    float nyj = ny[jg];
    float cmin = 1e30f;
#pragma unroll
    for (int r = 0; r < 8; ++r) {
      int ig = rowA + r + 8 * hi;
      float cost = 1.f - g[r] / (nx[ig] * nyj);
      float v = cost;
#pragma unroll
      for (int m = 1; m <= 8; m <<= 1) v = fminf(v, __shfl_xor(v, m, 32));
      if (jl == 0) atomicMin(&rowkey[ig], f2ord(v));
      cmin = fminf(cmin, cost);
    }
    cmin = fminf(cmin, __shfl_xor(cmin, 16, 32));
    if (hi == 0) atomicMin(&colkey[jg], f2ord(cmin));
  }
}

// ---------- moment loss covariance part: sum |cov_e - cov_s| fused ----------
__global__ void __launch_bounds__(256) k_cov(const bf16_t* __restrict__ xe,
                                             const bf16_t* __restrict__ xs,
                                             float* __restrict__ out_sum) {
  int lane = threadIdx.x & 31, wave = threadIdx.x >> 5;
  int i0 = blockIdx.y * 64, j0 = blockIdx.x * 64;
  int rowA = i0 + 16 * (wave & 3);
  int colB0 = j0 + 16 * ((wave >> 2) * 2), colB1 = colB0 + 16;
  v8f ae0 = {}, ae1 = {}, as0 = {}, as1 = {};
  for (int k0 = 0; k0 < N_DIM; k0 += 32) {
    v16bf aE = load_fragA(xe, rowA, N_DIM, k0, lane);
    v16bf aS = load_fragA(xs, rowA, N_DIM, k0, lane);
    v16bf bE0 = load_fragB(xe, colB0, N_DIM, k0, lane);
    v16bf bE1 = load_fragB(xe, colB1, N_DIM, k0, lane);
    v16bf bS0 = load_fragB(xs, colB0, N_DIM, k0, lane);
    v16bf bS1 = load_fragB(xs, colB1, N_DIM, k0, lane);
    ae0 = wmma_bf16(aE, bE0, ae0);
    ae1 = wmma_bf16(aE, bE1, ae1);
    as0 = wmma_bf16(aS, bS0, as0);
    as1 = wmma_bf16(aS, bS1, as1);
  }
  const float inv = 1.f / (float)(N_DIM - 1);
  float local = 0.f;
#pragma unroll
  for (int t = 0; t < 2; ++t) {
    const v8f& e = t ? ae1 : ae0;
    const v8f& s = t ? as1 : as0;
#pragma unroll
    for (int r = 0; r < 8; ++r) local += fabsf(e[r] * inv - s[r] * inv);
  }
  local = wave_sum(local);
  if (lane == 0) atomicAdd(out_sum, local);
}

// ---------- color REMD (d=3): fp32 WMMA 16x16x4, fused cost + mins ----------
__global__ void __launch_bounds__(256) k_color(const float* __restrict__ ux,
                                               const float* __restrict__ uy,
                                               const float* __restrict__ nx,
                                               const float* __restrict__ ny,
                                               unsigned* __restrict__ rowkey,
                                               unsigned* __restrict__ colkey) {
  int lane = threadIdx.x & 31, wave = threadIdx.x >> 5;
  int tile = blockIdx.x * 8 + wave;
  int n0 = (tile >> 8) * 16, m0 = (tile & 255) * 16;
  int hi = lane >> 4, l = lane & 15;
  // A 16x4 f32: elem e -> K = 2*hi + e ; B 4x16: same K split across halves
  const float* pa = ux + (size_t)(n0 + l) * 4 + 2 * hi;
  const float* pb = uy + (size_t)(m0 + l) * 4 + 2 * hi;
  v2f a; a[0] = pa[0]; a[1] = pa[1];
  v2f b; b[0] = pb[0]; b[1] = pb[1];
  v8f acc = {};
  acc = __builtin_amdgcn_wmma_f32_16x16x4_f32(false, a, false, b, (short)0, acc, false, false);
  int jg = m0 + l;
  float nyj = ny[jg];
  float cmin = 1e30f;
#pragma unroll
  for (int r = 0; r < 8; ++r) {
    int ig = n0 + r + 8 * hi;
    float nxi = nx[ig];
    float xy = acc[r];
    float e = nxi * nxi + nyj * nyj - 2.f * xy;
    e = fminf(fmaxf(e, 1e-5f), 1e5f);
    float cost = 1.f - xy / (nxi * nyj) + sqrtf(e * (1.f / 3.f));
    float v = cost;
#pragma unroll
    for (int m = 1; m <= 8; m <<= 1) v = fminf(v, __shfl_xor(v, m, 32));
    if (l == 0) atomicMin(&rowkey[ig], f2ord(v));
    cmin = fminf(cmin, cost);
  }
  cmin = fminf(cmin, __shfl_xor(cmin, 16, 32));
  if (hi == 0) atomicMin(&colkey[jg], f2ord(cmin));
}

// ---------- finalize ----------
__global__ void k_final(const unsigned* __restrict__ keys,  // rowmin|colmin|rowmin2|colmin2
                        const float* __restrict__ scalars,  // [0]=content_sum [1]=covdiff_sum
                        const float* __restrict__ me, const float* __restrict__ ms,
                        const int* __restrict__ alpha_ptr, float* __restrict__ out) {
  __shared__ float red[256];
  int t = threadIdx.x;
  float sums[5] = {0.f, 0.f, 0.f, 0.f, 0.f};
  for (int i = t; i < N_DIM; i += 256) {
    sums[0] += ord2f(keys[i]);
    sums[1] += ord2f(keys[N_DIM + i]);
    sums[2] += ord2f(keys[2 * N_DIM + i]);
    sums[3] += ord2f(keys[3 * N_DIM + i]);
  }
  for (int c = t; c < D_DIM; c += 256) sums[4] += fabsf(me[c] - ms[c]);
  float res[5];
  for (int s = 0; s < 5; ++s) {
    red[t] = sums[s];
    __syncthreads();
    for (int st = 128; st > 0; st >>= 1) {
      if (t < st) red[t] += red[t + st];
      __syncthreads();
    }
    res[s] = red[0];
    __syncthreads();
  }
  if (t == 0) {
    float alpha = (float)(*alpha_ptr);
    float lr = fmaxf(res[1] / (float)N_DIM, res[0] / (float)N_DIM);           // max(mean(min1), mean(min2))
    float lp = fmaxf(res[3] / (float)N_DIM, res[2] / (float)N_DIM) / alpha;   // color REMD / alpha
    float lm = res[4] / (float)D_DIM + scalars[1] / (float)(D_DIM * D_DIM);   // moment loss
    float lc = scalars[0] / ((float)N_DIM * (float)N_DIM);                    // content loss
    out[0] = (lr + lm + lp + lc * alpha) / (2.0f + alpha + 1.0f / alpha);
  }
}

extern "C" void kernel_launch(void* const* d_in, const int* in_sizes, int n_in,
                              void* d_out, int out_size, void* d_ws, size_t ws_size,
                              hipStream_t stream) {
  const float* ef = (const float*)d_in[0];
  const float* cf = (const float*)d_in[1];
  const float* sf = (const float*)d_in[2];
  const int* alpha = (const int*)d_in[3];

  char* w = (char*)d_ws;
  auto alloc = [&](size_t bytes) -> char* {
    char* p = w;
    w += (bytes + 255) & ~(size_t)255;
    return p;
  };
  bf16_t* Te = (bf16_t*)alloc(sizeof(bf16_t) * D_DIM * N_DIM);  // [N][512]
  bf16_t* Tc = (bf16_t*)alloc(sizeof(bf16_t) * D_DIM * N_DIM);
  bf16_t* Ts = (bf16_t*)alloc(sizeof(bf16_t) * D_DIM * N_DIM);
  bf16_t* Ec = (bf16_t*)alloc(sizeof(bf16_t) * D_DIM * N_DIM);  // centered [512][N]
  bf16_t* Sc = (bf16_t*)alloc(sizeof(bf16_t) * D_DIM * N_DIM);
  float* ne = (float*)alloc(sizeof(float) * N_DIM);
  float* nc = (float*)alloc(sizeof(float) * N_DIM);
  float* ns = (float*)alloc(sizeof(float) * N_DIM);
  float* me = (float*)alloc(sizeof(float) * D_DIM);
  float* ms = (float*)alloc(sizeof(float) * D_DIM);
  float* uvE = (float*)alloc(sizeof(float) * N_DIM * 4);
  float* uvS = (float*)alloc(sizeof(float) * N_DIM * 4);
  float* n3e = (float*)alloc(sizeof(float) * N_DIM);
  float* n3s = (float*)alloc(sizeof(float) * N_DIM);
  unsigned* keys = (unsigned*)alloc(sizeof(unsigned) * N_DIM * 4);
  float* scalars = (float*)alloc(sizeof(float) * 8);
  (void)ws_size; (void)in_sizes; (void)n_in; (void)out_size;

  unsigned* rowkey = keys;
  unsigned* colkey = keys + N_DIM;
  unsigned* rowkey2 = keys + 2 * N_DIM;
  unsigned* colkey2 = keys + 3 * N_DIM;

  k_init<<<1, 256, 0, stream>>>(keys, scalars);
  k_convert<<<N_DIM / 256, 256, 0, stream>>>(ef, Te, ne);
  k_convert<<<N_DIM / 256, 256, 0, stream>>>(cf, Tc, nc);
  k_convert<<<N_DIM / 256, 256, 0, stream>>>(sf, Ts, ns);
  k_chan_mean<<<D_DIM, 256, 0, stream>>>(ef, me);
  k_chan_mean<<<D_DIM, 256, 0, stream>>>(sf, ms);
  k_center<<<(D_DIM * N_DIM) / 256, 256, 0, stream>>>(ef, me, Ec);
  k_center<<<(D_DIM * N_DIM) / 256, 256, 0, stream>>>(sf, ms, Sc);
  k_yuv<<<N_DIM / 256, 256, 0, stream>>>(ef, uvE, n3e);
  k_yuv<<<N_DIM / 256, 256, 0, stream>>>(sf, uvS, n3s);

  k_content<<<dim3(N_DIM / 64, N_DIM / 64), 256, 0, stream>>>(Te, Tc, ne, nc, &scalars[0]);
  k_cost_min<<<dim3(N_DIM / 64, N_DIM / 64), 256, 0, stream>>>(Te, Ts, ne, ns, rowkey, colkey);
  k_cov<<<dim3(D_DIM / 64, D_DIM / 64), 256, 0, stream>>>(Ec, Sc, &scalars[1]);
  k_color<<<(N_DIM / 16) * (N_DIM / 16) / 8, 256, 0, stream>>>(uvE, uvS, n3e, n3s, rowkey2, colkey2);

  k_final<<<1, 256, 0, stream>>>(keys, scalars, me, ms, alpha, (float*)d_out);
}
